// TGCNCell_84155589198012
// MI455X (gfx1250) — matrix operand order repaired and checked by reading, not compiled, not run on hardware
//
#include <hip/hip_runtime.h>
#include <hip/hip_bf16.h>

typedef __attribute__((ext_vector_type(16))) __bf16 v16bf;
typedef __attribute__((ext_vector_type(8)))  __bf16 v8bf;
typedef __attribute__((ext_vector_type(8)))  float  v8f;

// ---- helpers ---------------------------------------------------------------

__device__ __forceinline__ __bf16 f2bf(float f) {
    union { float f; unsigned u; } v; v.f = f;
    unsigned r = (v.u + 0x7FFFu + ((v.u >> 16) & 1u)) >> 16;   // RNE
    unsigned short s = (unsigned short)r;
    return __builtin_bit_cast(__bf16, s);
}

// 16 contiguous bf16 (32B) as two b128 loads; works for LDS or global pointers
__device__ __forceinline__ v16bf load16(const __bf16* p) {
    v8bf lo = *(const v8bf*)p;
    v8bf hi = *(const v8bf*)(p + 8);
    return __builtin_shufflevector(lo, hi, 0, 1, 2, 3, 4, 5, 6, 7,
                                   8, 9, 10, 11, 12, 13, 14, 15);
}

// ---- degree / normalization ------------------------------------------------

__global__ void deg_init_kernel(float* __restrict__ deg, int n) {
    int i = blockIdx.x * blockDim.x + threadIdx.x;
    if (i < n) deg[i] = 1.0f;                       // self-loop
}

__global__ void deg_accum_kernel(const long long* __restrict__ ei,
                                 float* __restrict__ deg, int e) {
    int t = blockIdx.x * blockDim.x + threadIdx.x;
    if (t < e) {
        int d = (int)ei[(long long)e + t];          // dst row of edge_index
        atomicAdd(&deg[d], 1.0f);
    }
}

__global__ void dinv_kernel(float* __restrict__ deg, int n) {
    int i = blockIdx.x * blockDim.x + threadIdx.x;
    if (i < n) deg[i] = rsqrtf(deg[i]);             // deg >= 1 always
}

// ---- one-time W pre-pack into WMMA B-fragment order ------------------------
// Layout: Wp[((nb*12 + kb)*32 + lane)*16 + e] ; lane: n = nb*16 + (lane&15),
// half = lane>>4 ; element e: k = kb*32 + (e<8 ? half*8 + e : 16 + half*8 + e-8)   (K = 384 -> 12 kb)

template<int NOUT>
__global__ void pack_w_kernel(const float* __restrict__ W,
                              __bf16* __restrict__ Wp) {
    int t = blockIdx.x * blockDim.x + threadIdx.x;
    int lane = t & 31;
    int blk  = t >> 5;                  // nb*12 + kb
    int kb   = blk % 12;
    int nb   = blk / 12;
    if (nb >= NOUT / 16) return;
    int ncol = nb * 16 + (lane & 15);
    int half = lane >> 4;
    __bf16* dst = Wp + ((long long)blk * 32 + lane) * 16;
    #pragma unroll
    for (int e = 0; e < 16; ++e) {
        int k = kb * 32 + (e < 8 ? half * 8 + e : 16 + half * 8 + (e - 8));
        dst[e] = f2bf(W[(long long)k * NOUT + ncol]);
    }
}

// ---- WMMA GEMM: out[M,NOUT] = concat(x, (GATED? r*h : h)) @ W --------------
// BM=128 x BN=64 block, 8 waves, each wave owns a 32x32 macro-tile
// (2 A-frags x 2 B-frags -> 4 WMMAs per K-step). A staged in LDS in fragment
// order; B fragments loaded directly from pre-packed W (L2-resident).

template<int NOUT, bool GATED>
__global__ __launch_bounds__(256)
void gemm_wmma_kernel(const float* __restrict__ x,      // [n,128]
                      const float* __restrict__ h,      // [n,256]
                      const float* __restrict__ gates,  // [n,512] (GATED only)
                      const __bf16* __restrict__ Wp,    // packed W fragments
                      float* __restrict__ out,          // [n,NOUT]
                      int n) {
    constexpr int K = 384;
    constexpr int BM = 128, BN = 64, BK = 32;
    constexpr int AS = BK + 8;                 // 40 bf16 = 80B row stride (16B-mult)
    __shared__ __bf16 As[BM][AS];              // 10.0 KB, K-permuted fragment order

    const int block_m = blockIdx.x * BM;
    const int block_n = blockIdx.y * BN;
    const int tid  = threadIdx.x;
    const int lane = tid & 31;
    const int wave = tid >> 5;                 // 0..7
    const int tm   = (wave & 3) * 32;          // 32-row macro tile
    const int tn   = (wave >> 2) * 32;         // 32-col macro tile
    const int half = lane >> 4;                // wave32 lane half
    const int l16  = lane & 15;

    // packed-B base pointers for this wave's two 16-col tiles
    const long long nb0 = (long long)(block_n + tn) >> 4;
    const __bf16* wp0 = Wp + (nb0 * 12 * 32 + lane) * 16;   // advance 512/kb
    const __bf16* wp1 = wp0 + 12 * 512;                      // next nb

    v8f c00 = {}, c01 = {}, c10 = {}, c11 = {};

    for (int k0 = 0; k0 < K; k0 += BK) {
        // --- stage A (128x32): each thread: 1 row, 16 K-values, 2 packed stores
        {
            int row  = tid >> 1;               // 0..127
            int koff = (tid & 1) * 16;         // 0 or 16 (never straddles col 128)
            int gm   = block_m + row;
            int gk   = k0 + koff;
            float vals[16];
            if (gm < n) {
                if (gk < 128) {
                    const float* p = x + (long long)gm * 128 + gk;
                    #pragma unroll
                    for (int i = 0; i < 16; ++i) vals[i] = p[i];
                } else {
                    int hc = gk - 128;
                    const float* p = h + (long long)gm * 256 + hc;
                    #pragma unroll
                    for (int i = 0; i < 16; ++i) vals[i] = p[i];
                    if (GATED) {
                        const float* r = gates + (long long)gm * 512 + 256 + hc;
                        #pragma unroll
                        for (int i = 0; i < 16; ++i) vals[i] *= r[i];
                    }
                }
            } else {
                #pragma unroll
                for (int i = 0; i < 16; ++i) vals[i] = 0.0f;
            }
            // permute 8-chunks: chunk c of K -> LDS position ((c&1)<<4)|(((c>>1)&1)<<3)
            __align__(16) __bf16 t0[8], t1[8];
            #pragma unroll
            for (int i = 0; i < 8; ++i) { t0[i] = f2bf(vals[i]); t1[i] = f2bf(vals[8 + i]); }
            int c0p = koff >> 3;               // 0 or 2
            int p0 = ((c0p & 1) << 4) | (((c0p >> 1) & 1) << 3);
            int c1p = c0p + 1;                 // 1 or 3
            int p1 = ((c1p & 1) << 4) | (((c1p >> 1) & 1) << 3);
            *(uint4*)&As[row][p0] = *(const uint4*)t0;
            *(uint4*)&As[row][p1] = *(const uint4*)t1;
        }
        __syncthreads();

        // --- B fragments straight from packed W (global, L2-resident) ---
        int kb = k0 >> 5;
        v16bf b0 = load16(wp0 + (long long)kb * 512);
        v16bf b1 = load16(wp1 + (long long)kb * 512);
        // --- A fragments: contiguous b128 LDS loads ---
        v16bf a0 = load16(&As[tm + l16][half * 16]);
        v16bf a1 = load16(&As[tm + 16 + l16][half * 16]);

        c00 = __builtin_amdgcn_wmma_f32_16x16x32_bf16(false, a0, false, b0,
                                                      (short)0, c00, false, false);
        c01 = __builtin_amdgcn_wmma_f32_16x16x32_bf16(false, a0, false, b1,
                                                      (short)0, c01, false, false);
        c10 = __builtin_amdgcn_wmma_f32_16x16x32_bf16(false, a1, false, b0,
                                                      (short)0, c10, false, false);
        c11 = __builtin_amdgcn_wmma_f32_16x16x32_bf16(false, a1, false, b1,
                                                      (short)0, c11, false, false);
        __syncthreads();
    }

    // --- store D tiles (16x16 f32: lane half -> +8 rows, VGPR r -> row) ---
    #pragma unroll
    for (int r = 0; r < 8; ++r) {
        int gm0 = block_m + tm + half * 8 + r;
        int gm1 = gm0 + 16;
        long long col = (long long)block_n + tn + l16;
        if (gm0 < n) {
            out[(long long)gm0 * NOUT + col]      = c00[r];
            out[(long long)gm0 * NOUT + col + 16] = c01[r];
        }
        if (gm1 < n) {
            out[(long long)gm1 * NOUT + col]      = c10[r];
            out[(long long)gm1 * NOUT + col + 16] = c11[r];
        }
    }
}

// ---- aggregation: self-loop init, edge scatter, epilogues ------------------

template<int F>
__global__ void agg_selfloop_kernel(const float* __restrict__ xw,
                                    const float* __restrict__ dinv,
                                    float* __restrict__ agg, int n) {
    long long t = (long long)blockIdx.x * blockDim.x + threadIdx.x;
    long long total = (long long)n * F;
    if (t < total) {
        int node = (int)(t / F);
        float di = dinv[node];
        agg[t] = xw[t] * di * di;       // self-loop norm = dinv^2
    }
}

template<int F>
__global__ __launch_bounds__(F / 4)
void edge_scatter_kernel(const long long* __restrict__ ei,
                         const float* __restrict__ xw,
                         const float* __restrict__ dinv,
                         float* __restrict__ agg, int e) {
    int edge = blockIdx.x;
    if (edge >= e) return;
    int s = (int)ei[edge];
    int d = (int)ei[(long long)e + edge];
    float norm = dinv[s] * dinv[d];
    int f = threadIdx.x * 4;
    const float4 v = *(const float4*)(xw + (long long)s * F + f);
    float* o = agg + (long long)d * F + f;
    atomicAdd(o + 0, v.x * norm);
    atomicAdd(o + 1, v.y * norm);
    atomicAdd(o + 2, v.z * norm);
    atomicAdd(o + 3, v.w * norm);
}

__global__ void finalize_gates_kernel(float* __restrict__ gates,     // in-place on agg
                                      const float* __restrict__ bg,  // [512]
                                      long long total) {
    long long t = (long long)blockIdx.x * blockDim.x + threadIdx.x;
    if (t < total) {
        int f = (int)(t & 511);
        float v = gates[t] + bg[f];
        gates[t] = 1.0f / (1.0f + __expf(-v));
    }
}

__global__ void finalize_h_kernel(const float* __restrict__ agg2,   // [n,256]
                                  const float* __restrict__ bu,     // [256]
                                  const float* __restrict__ gates,  // [n,512]
                                  const float* __restrict__ h,      // [n,256]
                                  float* __restrict__ out, int n) {
    long long t = (long long)blockIdx.x * blockDim.x + threadIdx.x;
    long long total = (long long)n * 256;
    if (t < total) {
        int node = (int)(t >> 8);
        int f    = (int)(t & 255);
        float z  = gates[(long long)node * 512 + f];          // first half = z
        float ht = tanhf(agg2[t] + bu[f]);
        out[t] = (1.0f - z) * h[t] + z * ht;
    }
}

// ---- launcher --------------------------------------------------------------

extern "C" void kernel_launch(void* const* d_in, const int* in_sizes, int n_in,
                              void* d_out, int out_size, void* d_ws, size_t ws_size,
                              hipStream_t stream) {
    const float*     x   = (const float*)d_in[0];        // [n,128]
    const long long* ei  = (const long long*)d_in[1];    // [2,e] int64
    const float*     h   = (const float*)d_in[2];        // [n,256]
    const float*     Wg  = (const float*)d_in[3];        // [384,512]
    const float*     bg  = (const float*)d_in[4];        // [512]
    const float*     Wu  = (const float*)d_in[5];        // [384,256]
    const float*     bu  = (const float*)d_in[6];        // [256]
    float*           out = (float*)d_out;                // [n,256]

    const int n = in_sizes[0] / 128;
    const int e = in_sizes[1] / 2;

    // workspace (floats): [xw_gate N*512][gates N*512][dinv N][WpG][WpU]
    // xw_update / agg_update reuse the xw_gate region after gates are done.
    float* ws       = (float*)d_ws;
    const long long N512 = (long long)n * 512;
    const long long N256 = (long long)n * 256;
    float*  xw_gate = ws;
    float*  gates   = ws + N512;
    float*  dinv    = ws + 2 * N512;
    __bf16* WpG     = (__bf16*)(ws + 2 * N512 + n);       // 384*512 bf16
    __bf16* WpU     = WpG + 384 * 512;                    // 384*256 bf16
    float*  xw_upd  = ws;                 // reuse
    float*  agg_upd = ws + N256;          // reuse (disjoint from xw_upd)

    const int mb = (n + 127) / 128;

    // 0) pre-pack weights into WMMA fragment order (one-time, tiny)
    pack_w_kernel<512><<<(32 * 12 * 32 + 255) / 256, 256, 0, stream>>>(Wg, WpG);
    pack_w_kernel<256><<<(16 * 12 * 32 + 255) / 256, 256, 0, stream>>>(Wu, WpU);

    // 1) degrees -> dinv
    deg_init_kernel<<<(n + 255) / 256, 256, 0, stream>>>(dinv, n);
    deg_accum_kernel<<<(e + 255) / 256, 256, 0, stream>>>(ei, dinv, e);
    dinv_kernel<<<(n + 255) / 256, 256, 0, stream>>>(dinv, n);

    // 2) GEMM1: concat(x,h) @ W_gate -> xw_gate
    gemm_wmma_kernel<512, false><<<dim3(mb, 8), 256, 0, stream>>>(
        x, h, nullptr, WpG, xw_gate, n);

    // 3) aggregate + bias + sigmoid -> gates
    agg_selfloop_kernel<512><<<(unsigned)((N512 + 255) / 256), 256, 0, stream>>>(
        xw_gate, dinv, gates, n);
    edge_scatter_kernel<512><<<e, 128, 0, stream>>>(ei, xw_gate, dinv, gates, e);
    finalize_gates_kernel<<<(unsigned)((N512 + 255) / 256), 256, 0, stream>>>(
        gates, bg, N512);

    // 4) GEMM2: concat(x, r*h) @ W_update -> xw_upd
    gemm_wmma_kernel<256, true><<<dim3(mb, 4), 256, 0, stream>>>(
        x, h, gates, WpU, xw_upd, n);

    // 5) aggregate, then h_new = (1-z)*h + z*tanh(agg + b_update)
    agg_selfloop_kernel<256><<<(unsigned)((N256 + 255) / 256), 256, 0, stream>>>(
        xw_upd, dinv, agg_upd, n);
    edge_scatter_kernel<256><<<e, 64, 0, stream>>>(ei, xw_upd, dinv, agg_upd, e);
    finalize_h_kernel<<<(unsigned)((N256 + 255) / 256), 256, 0, stream>>>(
        agg_upd, bu, gates, h, out, n);
}
